// Net_GAT_ASAP_GlobalPooling_56727928046280
// MI455X (gfx1250) — matrix operand order repaired and checked by reading, not compiled, not run on hardware
//
#include <hip/hip_runtime.h>
#include <hip/hip_bf16.h>

// ---------------------------------------------------------------------------
// MI455X (gfx1250, wave32) implementation of the GATv2+ASAP GNN reference.
// All dense matmuls go through v_wmma_f32_16x16x32_bf16 (f32 accumulate).
// GEMM: 32x32 output per wave (2x2 register blocking -> 4 WMMA per K-step,
// fragments reused, 16 loads/WMMA), branchless main loop + select-based tail.
// ---------------------------------------------------------------------------

typedef __attribute__((ext_vector_type(16))) __bf16 bf16x16;
typedef __attribute__((ext_vector_type(8)))  float  f32x8;

#define NEG_SLOPE 0.2f
#define BN_EPS    1e-5f

#define WMMA_BF16(Afrag, Bfrag, Cacc) \
    __builtin_amdgcn_wmma_f32_16x16x32_bf16(false, (Afrag), false, (Bfrag), \
                                            (short)0, (Cacc), false, false)

// ------------------------- WMMA GEMM ---------------------------------------
// C[M,N] = op(A)[M,K] @ B[K,N] (+ bias[N])
// TRANSA==0: A[m*lda+k]   TRANSA==1: A[k*lda+m]
// Fragment layouts per CDNA5 ISA 7.12.2 (16-bit A 16x32, B 32x16, C f32).
template <int TRANSA>
__global__ __launch_bounds__(128)
void k_gemm_bf16(const float* __restrict__ A, const float* __restrict__ B,
                 const float* __restrict__ bias, float* __restrict__ C,
                 int M, int N, int K, int lda, int ldb, int ldc)
{
    int wave = threadIdx.x >> 5;
    int lane = threadIdx.x & 31;
    int tilesN = (N + 31) >> 5;
    int tilesM = (M + 31) >> 5;
    int tile = blockIdx.x * 4 + wave;
    if (tile >= tilesM * tilesN) return;
    int mt = tile / tilesN;
    int nt = tile - mt * tilesN;

    int mrow = lane & 15;
    int hi   = lane >> 4;

    int mg0 = mt * 32 + mrow, mg1 = mg0 + 16;   // A rows this lane supplies
    int ng0 = nt * 32 + mrow, ng1 = ng0 + 16;   // B cols this lane supplies
    int mc0 = (mg0 < M) ? mg0 : (M - 1);        // clamp -> always-valid loads
    int mc1 = (mg1 < M) ? mg1 : (M - 1);
    int nc0 = (ng0 < N) ? ng0 : (N - 1);
    int nc1 = (ng1 < N) ? ng1 : (N - 1);

    const float* pa0 = TRANSA ? (A + mc0) : (A + (long)mc0 * lda);
    const float* pa1 = TRANSA ? (A + mc1) : (A + (long)mc1 * lda);
    const float* pb0 = B + nc0;
    const float* pb1 = B + nc1;

    f32x8 acc00 = {}, acc01 = {}, acc10 = {}, acc11 = {};
    int Kfull = K & ~31;
    for (int k0 = 0; k0 < Kfull; k0 += 32) {
        bf16x16 a0, a1, b0, b1;
#pragma unroll
        for (int v = 0; v < 8; ++v) {
            int kk = ((v < 4) ? 0 : 16) + hi * 8 + 2 * (v & 3);
            int ka = k0 + kk;
            float x0, x1, y0, y1;
            if (TRANSA) {
                x0 = pa0[(long)ka * lda];       x1 = pa0[(long)(ka + 1) * lda];
                y0 = pa1[(long)ka * lda];       y1 = pa1[(long)(ka + 1) * lda];
            } else {
                x0 = pa0[ka];                   x1 = pa0[ka + 1];
                y0 = pa1[ka];                   y1 = pa1[ka + 1];
            }
            a0[2 * v] = (__bf16)x0;  a0[2 * v + 1] = (__bf16)x1;
            a1[2 * v] = (__bf16)y0;  a1[2 * v + 1] = (__bf16)y1;

            int kb = k0 + kk;
            b0[2 * v]     = (__bf16)pb0[(long)kb * ldb];
            b0[2 * v + 1] = (__bf16)pb0[(long)(kb + 1) * ldb];
            b1[2 * v]     = (__bf16)pb1[(long)kb * ldb];
            b1[2 * v + 1] = (__bf16)pb1[(long)(kb + 1) * ldb];
        }
        acc00 = WMMA_BF16(a0, b0, acc00);
        acc01 = WMMA_BF16(a0, b1, acc01);
        acc10 = WMMA_BF16(a1, b0, acc10);
        acc11 = WMMA_BF16(a1, b1, acc11);
    }
    if (Kfull < K) {   // K tail: clamped addresses + selects, no branches
        bf16x16 a0, a1, b0, b1;
#pragma unroll
        for (int v = 0; v < 8; ++v) {
            int kk = ((v < 4) ? 0 : 16) + hi * 8 + 2 * (v & 3);
            int ka = Kfull + kk;
            int ka0 = (ka < K) ? ka : (K - 1);
            int ka1 = (ka + 1 < K) ? (ka + 1) : (K - 1);
            float x0, x1, y0, y1;
            if (TRANSA) {
                x0 = pa0[(long)ka0 * lda];      x1 = pa0[(long)ka1 * lda];
                y0 = pa1[(long)ka0 * lda];      y1 = pa1[(long)ka1 * lda];
            } else {
                x0 = pa0[ka0];                  x1 = pa0[ka1];
                y0 = pa1[ka0];                  y1 = pa1[ka1];
            }
            x0 = (ka < K) ? x0 : 0.f;  x1 = (ka + 1 < K) ? x1 : 0.f;
            y0 = (ka < K) ? y0 : 0.f;  y1 = (ka + 1 < K) ? y1 : 0.f;
            a0[2 * v] = (__bf16)x0;  a0[2 * v + 1] = (__bf16)x1;
            a1[2 * v] = (__bf16)y0;  a1[2 * v + 1] = (__bf16)y1;

            int kb = Kfull + kk;
            int kb0 = (kb < K) ? kb : (K - 1);
            int kb1 = (kb + 1 < K) ? (kb + 1) : (K - 1);
            float u0 = pb0[(long)kb0 * ldb];
            float u1 = pb0[(long)kb1 * ldb];
            float w0 = pb1[(long)kb0 * ldb];
            float w1 = pb1[(long)kb1 * ldb];
            u0 = (kb < K) ? u0 : 0.f;  u1 = (kb + 1 < K) ? u1 : 0.f;
            w0 = (kb < K) ? w0 : 0.f;  w1 = (kb + 1 < K) ? w1 : 0.f;
            b0[2 * v] = (__bf16)u0;  b0[2 * v + 1] = (__bf16)u1;
            b1[2 * v] = (__bf16)w0;  b1[2 * v + 1] = (__bf16)w1;
        }
        acc00 = WMMA_BF16(a0, b0, acc00);
        acc01 = WMMA_BF16(a0, b1, acc01);
        acc10 = WMMA_BF16(a1, b0, acc10);
        acc11 = WMMA_BF16(a1, b1, acc11);
    }

    int cg0 = nt * 32 + mrow, cg1 = cg0 + 16;
    float bv0 = (bias && cg0 < N) ? bias[cg0] : 0.f;
    float bv1 = (bias && cg1 < N) ? bias[cg1] : 0.f;
#pragma unroll
    for (int v = 0; v < 8; ++v) {
        int r0 = mt * 32 + hi * 8 + v;
        int r1 = r0 + 16;
        if (r0 < M && cg0 < N) C[(long)r0 * ldc + cg0] = acc00[v] + bv0;
        if (r0 < M && cg1 < N) C[(long)r0 * ldc + cg1] = acc01[v] + bv1;
        if (r1 < M && cg0 < N) C[(long)r1 * ldc + cg0] = acc10[v] + bv0;
        if (r1 < M && cg1 < N) C[(long)r1 * ldc + cg1] = acc11[v] + bv1;
    }
}

// ------------------------- small utility kernels ---------------------------
__global__ void k_zero(float* p, int n) {
    int i = blockIdx.x * blockDim.x + threadIdx.x;
    if (i < n) p[i] = 0.f;
}

__global__ void k_scatter_adj(const int* __restrict__ ei, float* __restrict__ A,
                              int nE, int N) {
    int k = blockIdx.x * blockDim.x + threadIdx.x;
    if (k < nE) A[(long)ei[k] * N + ei[nE + k]] = 1.0f;
}

__global__ void k_fix_diag(float* A, int N) {   // A += diag(diag==0 ? 1 : 0)
    int i = blockIdx.x * blockDim.x + threadIdx.x;
    if (i < N && A[(long)i * N + i] == 0.f) A[(long)i * N + i] = 1.0f;
}

__global__ void k_set_diag1(float* A, int N) {
    int i = blockIdx.x * blockDim.x + threadIdx.x;
    if (i < N) A[(long)i * N + i] = 1.0f;
}

// e[h][j][i] = mask(j,i) ? sum_c lrelu(gl[j,h,c]+gr[i,h,c])*att[h,c] : -inf
__global__ void k_gat_scores(const float* __restrict__ gl, const float* __restrict__ gr,
                             const float* __restrict__ att, const float* __restrict__ A,
                             float* __restrict__ e, int N, int H, int C)
{
    long idx = (long)blockIdx.x * blockDim.x + threadIdx.x;
    long total = (long)H * N * N;
    if (idx >= total) return;
    int i = (int)(idx % N);
    long t = idx / N;
    int j = (int)(t % N);
    int h = (int)(t / N);
    float out;
    if (A[(long)j * N + i] != 0.f || j == i) {
        const float* pl = gl + ((long)j * H + h) * C;
        const float* pr = gr + ((long)i * H + h) * C;
        const float* pa = att + (long)h * C;
        float acc = 0.f;
        for (int c = 0; c < C; ++c) {
            float z = pl[c] + pr[c];
            z = (z > 0.f) ? z : NEG_SLOPE * z;
            acc += z * pa[c];
        }
        out = acc;
    } else {
        out = -__builtin_inff();
    }
    e[((long)h * N + j) * N + i] = out;
}

// softmax over rows (axis 0) for H stacked NxN matrices, in place
__global__ void k_softmax_cols(float* e, int N, int H) {
    int cid = blockIdx.x * blockDim.x + threadIdx.x;
    if (cid >= H * N) return;
    int h = cid / N, i = cid - h * N;
    float* p = e + (long)h * N * N + i;
    float m = -__builtin_inff();
    for (int j = 0; j < N; ++j) m = fmaxf(m, p[(long)j * N]);
    float s = 0.f;
    for (int j = 0; j < N; ++j) {
        float v = expf(p[(long)j * N] - m);
        p[(long)j * N] = v;
        s += v;
    }
    float inv = 1.f / s;
    for (int j = 0; j < N; ++j) p[(long)j * N] *= inv;
}

// per-feature mean/var of relu(y) over N rows (one block per feature)
__global__ void k_bn_stats(const float* __restrict__ y, float* mu, float* var,
                           int N, int D) {
    int d = blockIdx.x;
    __shared__ float s1[256], s2[256];
    int t = threadIdx.x;
    float a = 0.f, b = 0.f;
    for (int n = t; n < N; n += 256) {
        float v = fmaxf(y[(long)n * D + d], 0.f);
        a += v; b += v * v;
    }
    s1[t] = a; s2[t] = b;
    __syncthreads();
    for (int s = 128; s > 0; s >>= 1) {
        if (t < s) { s1[t] += s1[t + s]; s2[t] += s2[t + s]; }
        __syncthreads();
    }
    if (t == 0) {
        float m = s1[0] / (float)N;
        mu[d] = m;
        var[d] = s2[0] / (float)N - m * m;
    }
}

__global__ void k_bn_apply(float* y, const float* __restrict__ mu,
                           const float* __restrict__ var,
                           const float* __restrict__ bnb,
                           const float* __restrict__ bng, int N, int D) {
    long idx = (long)blockIdx.x * blockDim.x + threadIdx.x;
    if (idx >= (long)N * D) return;
    int d = (int)(idx % D);
    float v = fmaxf(y[idx], 0.f);
    y[idx] = (v - mu[d]) * rsqrtf(var[d] + BN_EPS) * bng[d] + bnb[d];
}

// xq[v][d] = max over u with A[u][v]!=0 of x[u][d] (else finfo.min)
__global__ void k_masked_colmax(const float* __restrict__ x, const float* __restrict__ A,
                                float* __restrict__ xq, int N, int D) {
    long idx = (long)blockIdx.x * blockDim.x + threadIdx.x;
    if (idx >= (long)N * D) return;
    int v = (int)(idx / D);
    int d = (int)(idx % D);
    float m = -3.4028234663852886e38f;
    for (int u = 0; u < N; ++u)
        if (A[(long)u * N + v] != 0.f) m = fmaxf(m, x[(long)u * D + d]);
    xq[idx] = m;
}

__global__ void k_matvec(const float* __restrict__ X, const float* __restrict__ w,
                         const float* __restrict__ b, float* __restrict__ out,
                         int N, int D) {
    int n = blockIdx.x * blockDim.x + threadIdx.x;
    if (n >= N) return;
    float s = b ? b[0] : 0.f;
    const float* p = X + (long)n * D;
    for (int d = 0; d < D; ++d) s += p[d] * w[d];
    out[n] = s;
}

__global__ void k_asap_score(const float* __restrict__ sp, const float* __restrict__ sq,
                             const float* __restrict__ attb, const float* __restrict__ A,
                             float* __restrict__ sc, int N) {
    long idx = (long)blockIdx.x * blockDim.x + threadIdx.x;
    if (idx >= (long)N * N) return;
    int u = (int)(idx / N);
    int v = (int)(idx % N);
    if (A[idx] != 0.f) {
        float z = sp[u] + sq[v] + attb[0];
        sc[idx] = (z > 0.f) ? z : NEG_SLOPE * z;
    } else {
        sc[idx] = -__builtin_inff();
    }
}

// fit[v] = sigmoid( sum_u mask*a_msg[u] - deg_in[v]*b_msg[v] + le3[v] )
__global__ void k_fitness(const float* __restrict__ A, const float* __restrict__ am,
                          const float* __restrict__ bm, const float* __restrict__ l3,
                          float* __restrict__ fit, int N) {
    int v = blockIdx.x * blockDim.x + threadIdx.x;
    if (v >= N) return;
    float s = 0.f, deg = 0.f;
    for (int u = 0; u < N; ++u)
        if (A[(long)u * N + v] != 0.f) { s += am[u]; deg += 1.f; }
    float x = s - deg * bm[v] + l3[v];
    fit[v] = 1.f / (1.f + expf(-x));
}

// top-K (sorted, stable ties -> lower index), single block of 512 threads
__global__ void k_topk(const float* __restrict__ fit, float* __restrict__ fitk,
                       int* __restrict__ perm, int N, int K) {
    __shared__ float vals[512];
    __shared__ float rv[512];
    __shared__ int   ri[512];
    int t = threadIdx.x;
    vals[t] = (t < N) ? fit[t] : -__builtin_inff();
    __syncthreads();
    for (int k = 0; k < K; ++k) {
        rv[t] = vals[t]; ri[t] = t;
        __syncthreads();
        for (int s = 256; s > 0; s >>= 1) {
            if (t < s) {
                float a = rv[t], b = rv[t + s];
                if (b > a || (b == a && ri[t + s] < ri[t])) {
                    rv[t] = b; ri[t] = ri[t + s];
                }
            }
            __syncthreads();
        }
        if (t == 0) {
            perm[k] = ri[0];
            fitk[k] = rv[0];
            vals[ri[0]] = -__builtin_inff();
        }
        __syncthreads();
    }
}

__global__ void k_gather_scale(const float* __restrict__ xn, const int* __restrict__ perm,
                               const float* __restrict__ fitk, float* __restrict__ xo,
                               int K, int D) {
    long idx = (long)blockIdx.x * blockDim.x + threadIdx.x;
    if (idx >= (long)K * D) return;
    int r = (int)(idx / D);
    int d = (int)(idx % D);
    xo[idx] = xn[(long)perm[r] * D + d] * fitk[r];
}

__global__ void k_gather_cols(const float* __restrict__ alpha, const int* __restrict__ perm,
                              float* __restrict__ S, int N, int K) {
    long idx = (long)blockIdx.x * blockDim.x + threadIdx.x;
    if (idx >= (long)N * K) return;
    int u = (int)(idx / K);
    int r = (int)(idx % K);
    S[idx] = alpha[(long)u * N + perm[r]];
}

__global__ void k_mean_rows(const float* __restrict__ x, float* __restrict__ g,
                            int N, int D) {
    int d = blockIdx.x * blockDim.x + threadIdx.x;
    if (d >= D) return;
    float s = 0.f;
    for (int n = 0; n < N; ++n) s += x[(long)n * D + d];
    g[d] = s / (float)N;
}

__global__ void k_fc(const float* __restrict__ in, const float* __restrict__ W,
                     const float* __restrict__ b, float* __restrict__ out,
                     int D, int O) {
    int o = blockIdx.x * blockDim.x + threadIdx.x;
    if (o >= O) return;
    float s = b[o];
    for (int d = 0; d < D; ++d) s += in[d] * W[(long)d * O + o];
    out[o] = s;
}

// ------------------------- host orchestration ------------------------------
extern "C" void kernel_launch(void* const* d_in, const int* in_sizes, int n_in,
                              void* d_out, int out_size, void* d_ws, size_t ws_size,
                              hipStream_t stream)
{
    (void)in_sizes; (void)n_in; (void)out_size; (void)ws_size;
    const float* x_in = (const float*)d_in[0];
    const int*   ei   = (const int*)d_in[1];
    // d_in[2] = batch (unused, single graph)
    auto prm = [&](int i) -> const float* { return (const float*)d_in[3 + i]; };
    // Param leaf layout per block (jax pytree, dict keys sorted):
    //  0 att_b | 1 att_w | 2 le1.W | 3 le1.b | 4 le2W | 5 le3.W | 6 le3.b
    //  7 lin.W | 8 lin.b | 9..14 bn{b,g}x3 | 15+6c: Wl,Wr,att,bias,bl,br
    // fc1.W=99 fc1.b=100 fc2.W=101 fc2.b=102

    char* wp = (char*)d_ws;
    auto allocf = [&](size_t n) -> float* {
        float* r = (float*)wp;
        wp += ((n * sizeof(float) + 255) & ~(size_t)255);
        return r;
    };
    float* A0  = allocf(512 * 512);
    float* A1  = allocf(512 * 512);
    float* XA  = allocf(512 * 768);
    float* XB  = allocf(512 * 768);
    float* GL  = allocf(512 * 768);
    float* GR  = allocf(512 * 768);
    float* E   = allocf(3 * 512 * 512);
    float* XQ0 = allocf(512 * 256);
    float* XQ1 = allocf(512 * 256);
    float* SC  = allocf(512 * 512);
    float* XN  = allocf(512 * 256);
    float* Sm  = allocf(512 * 512);
    float* Tm  = allocf(512 * 512);
    float* MU  = allocf(768);
    float* VA  = allocf(768);
    float* SP  = allocf(512);
    float* SQv = allocf(512);
    float* AM  = allocf(512);
    float* BM  = allocf(512);
    float* L3  = allocf(512);
    float* FIT = allocf(512);
    float* FTK = allocf(512);
    int*   PRM = (int*)allocf(512);
    float* GV  = allocf(256);
    float* HV  = allocf(256);

    auto gemm = [&](const float* A, const float* B, const float* bias, float* C,
                    int M, int N, int K, int lda, int ldb, int ldc, int tA) {
        int tiles = ((M + 31) / 32) * ((N + 31) / 32);
        int blocks = (tiles + 3) / 4;
        if (tA)
            k_gemm_bf16<1><<<blocks, 128, 0, stream>>>(A, B, bias, C, M, N, K, lda, ldb, ldc);
        else
            k_gemm_bf16<0><<<blocks, 128, 0, stream>>>(A, B, bias, C, M, N, K, lda, ldb, ldc);
    };

    // --- build dense adjacency A0 from edge_index ---
    k_zero<<<(512 * 512 + 255) / 256, 256, 0, stream>>>(A0, 512 * 512);
    k_scatter_adj<<<(8192 + 255) / 256, 256, 0, stream>>>(ei, A0, 8192, 512);

    const int Ns[3]   = {512, 359, 252};   // nodes entering each block
    const int Kp[3]   = {359, 252, 177};   // ceil(0.7*n)
    const int INs[3]  = {39, 128, 128};
    const int HIDs[3] = {64, 128, 256};
    const int OUTs[3] = {128, 128, 256};
    float* Abufs[2] = {A0, A1};

    const float* xc = x_in;
    int aidx = 0;
    for (int b = 0; b < 3; ++b) {
        int base = b * 33;
        int n = Ns[b];
        float* Acur  = Abufs[aidx & 1];
        float* Anext = Abufs[(aidx + 1) & 1];
        int fin = INs[b];
        float* ybufs[3] = {XB, XA, XB};

        // --- 3 GATv2 convs, each: proj -> scores -> softmax -> agg -> relu+BN
        for (int c = 0; c < 3; ++c) {
            int H = (c == 2) ? 1 : 3;
            int C = (c == 2) ? OUTs[b] : HIDs[b];
            int D = H * C;
            int cb = base + 15 + 6 * c;
            gemm(xc, prm(cb + 0), prm(cb + 4), GL, n, D, fin, fin, D, D, 0); // x@Wl+bl
            gemm(xc, prm(cb + 1), prm(cb + 5), GR, n, D, fin, fin, D, D, 0); // x@Wr+br
            long tot = (long)H * n * n;
            k_gat_scores<<<(int)((tot + 255) / 256), 256, 0, stream>>>(
                GL, GR, prm(cb + 2), Acur, E, n, H, C);
            k_softmax_cols<<<(H * n + 127) / 128, 128, 0, stream>>>(E, n, H);
            float* y = ybufs[c];
            for (int h = 0; h < H; ++h)  // out_h = alpha_h^T @ gl_h + bias_h
                gemm(E + (long)h * n * n, GL + h * C, prm(cb + 3) + h * C,
                     y + h * C, n, C, n, n, D, D, 1);
            k_bn_stats<<<D, 256, 0, stream>>>(y, MU, VA, n, D);
            k_bn_apply<<<(int)(((long)n * D + 255) / 256), 256, 0, stream>>>(
                y, MU, VA, prm(base + 9 + 2 * c), prm(base + 10 + 2 * c), n, D);
            xc = y;
            fin = D;
        }

        // --- ASAP pooling ---
        int d = OUTs[b], Kb = Kp[b];
        k_fix_diag<<<(n + 255) / 256, 256, 0, stream>>>(Acur, n);
        k_masked_colmax<<<(int)(((long)n * d + 255) / 256), 256, 0, stream>>>(
            xc, Acur, XQ0, n, d);
        gemm(XQ0, prm(base + 7), prm(base + 8), XQ1, n, d, d, d, d, d, 0); // xq@linW+b
        k_matvec<<<(n + 127) / 128, 128, 0, stream>>>(XQ1, prm(base + 1), nullptr, SQv, n, d);
        k_matvec<<<(n + 127) / 128, 128, 0, stream>>>(xc, prm(base + 1) + d, nullptr, SP, n, d);
        k_asap_score<<<(int)(((long)n * n + 255) / 256), 256, 0, stream>>>(
            SP, SQv, prm(base + 0), Acur, SC, n);
        k_softmax_cols<<<(n + 127) / 128, 128, 0, stream>>>(SC, n, 1);
        gemm(SC, xc, nullptr, XN, n, d, n, n, d, d, 1);   // x_new = alpha^T @ x
        k_matvec<<<(n + 127) / 128, 128, 0, stream>>>(XN, prm(base + 2), prm(base + 3), AM, n, d);
        k_matvec<<<(n + 127) / 128, 128, 0, stream>>>(XN, prm(base + 4), nullptr, BM, n, d);
        k_matvec<<<(n + 127) / 128, 128, 0, stream>>>(XN, prm(base + 5), prm(base + 6), L3, n, d);
        k_fitness<<<(n + 127) / 128, 128, 0, stream>>>(Acur, AM, BM, L3, FIT, n);
        k_topk<<<1, 512, 0, stream>>>(FIT, FTK, PRM, n, Kb);
        k_gather_scale<<<(int)(((long)Kb * d + 255) / 256), 256, 0, stream>>>(
            XN, PRM, FTK, XA, Kb, d);                      // x_out -> XA
        if (b < 2) {                                       // A_new = S^T A S (unused after b3)
            k_gather_cols<<<(int)(((long)n * Kb + 255) / 256), 256, 0, stream>>>(
                SC, PRM, Sm, n, Kb);
            gemm(Acur, Sm, nullptr, Tm, n, Kb, n, n, Kb, Kb, 0);      // T = A @ S
            gemm(Sm, Tm, nullptr, Anext, Kb, Kb, n, Kb, Kb, Kb, 1);   // S^T @ T
            k_set_diag1<<<(Kb + 255) / 256, 256, 0, stream>>>(Anext, Kb);
        }
        xc = XA;
        aidx++;
    }

    // --- global mean pool + MLP head ---
    k_mean_rows<<<(256 + 63) / 64, 64, 0, stream>>>(xc, GV, 177, 256);
    k_fc<<<(256 + 63) / 64, 64, 0, stream>>>(GV, prm(99), prm(100), HV, 256, 256);
    k_fc<<<1, 64, 0, stream>>>(HV, prm(101), prm(102), (float*)d_out, 256, 10);
}
